// CrossAttention_58806692217094
// MI455X (gfx1250) — matrix operand order repaired and checked by reading, not compiled, not run on hardware
//
#include <hip/hip_runtime.h>

typedef _Float16 h16;
typedef _Float16 v4h  __attribute__((ext_vector_type(4)));
typedef _Float16 v8h  __attribute__((ext_vector_type(8)));
typedef _Float16 v16h __attribute__((ext_vector_type(16)));
typedef float    v8f  __attribute__((ext_vector_type(8)));
typedef unsigned int u32x4 __attribute__((ext_vector_type(4)));
typedef int          i32x4 __attribute__((ext_vector_type(4)));
typedef int          i32x8 __attribute__((ext_vector_type(8)));

// Problem dims (fixed by the reference)
constexpr int NB = 8, NT = 1024, NS = 1024, NE = 768, NH = 768;

// Raw split workgroup barrier: execution sync WITHOUT the acq/rel counter
// drain __syncthreads() carries. LDS producer ordering is handled separately
// by s_wait_tensorcnt before the signal.
__device__ __forceinline__ void wg_barrier() {
  asm volatile("s_barrier_signal -1\n\ts_barrier_wait -1" ::: "memory");
}

// ---------------- WMMA A-fragment loader (contiguous k order, same as B) ------
// A: 16x32 f16 (MxK). lanes0-15: row M=lane, halves0-15 = K+0..15;
// lanes16-31: same rows, K offset +16.
__device__ __forceinline__ v16h frag_a(const h16* __restrict__ base, int lda,
                                       int row, int k0, int lane) {
  int kb = (lane & 16) ? 16 : 0;
  const h16* p = base + (size_t)row * lda + k0 + kb;
  v8h lo = *(const v8h*)p;
  v8h hi = *(const v8h*)(p + 8);
  return __builtin_shufflevector(lo, hi, 0,1,2,3,4,5,6,7,8,9,10,11,12,13,14,15);
}

// B fragment (32x16 KxN) out of an LDS tile stored [N-row][K] row-major:
// lane = column N (tile row), 16 consecutive K halves (lanes16-31 at K+16).
__device__ __forceinline__ v16h frag_b_lds(const h16* bp) {
  v8h lo = *(const v8h*)bp;
  v8h hi = *(const v8h*)(bp + 8);
  return __builtin_shufflevector(lo, hi, 0,1,2,3,4,5,6,7,8,9,10,11,12,13,14,15);
}

// ---------------- TDM: DMA one 64(N-rows) x 64(K) f16 tile into LDS ----------
// D# per ISA 8.3/8.4: 2D tile, data_size=2B, tile_dim0=64 (contiguous K),
// tile_dim1=64 (rows), tensor_dim0_stride=ldb elements. LDS receives the tile
// row-major [64][64] halves, the B-fragment feed layout.
__device__ __forceinline__ void tdm_load_tile(unsigned lds_byte,
                                              const h16* gptr, int ldb) {
  unsigned long long ga = (unsigned long long)(size_t)gptr;
  u32x4 g0;
  g0[0] = 1u;                                   // count=1, user descriptor
  g0[1] = lds_byte;                             // LDS byte address
  g0[2] = (unsigned)(ga & 0xFFFFFFFFu);         // global_addr[31:0]
  g0[3] = (unsigned)((ga >> 32) & 0x01FFFFFFu)  // global_addr[56:32]
          | (2u << 30);                         // type=2 (image)
  i32x8 g1;
  g1[0] = 0x10000;        // workgroup_mask=0, data_size=1 (2 bytes)
  g1[1] = 64 << 16;       // tensor_dim0[15:0] = 64
  g1[2] = 64 << 16;       // tensor_dim0[31:16]=0, tensor_dim1[15:0] = 64
  g1[3] = 64 << 16;       // tensor_dim1[31:16]=0, tile_dim0 = 64
  g1[4] = 64;             // tile_dim1 = 64, tile_dim2 = 0
  g1[5] = ldb;            // tensor_dim0_stride[31:0]
  g1[6] = 0;              // stride[47:32]=0, tensor_dim1_stride[15:0]=0
  g1[7] = 0;
  i32x4 z4 = {0, 0, 0, 0};
#if __clang_major__ >= 23
  i32x8 z8 = {0, 0, 0, 0, 0, 0, 0, 0};
  __builtin_amdgcn_tensor_load_to_lds(g0, g1, z4, z4, z8, 0);
#else
  __builtin_amdgcn_tensor_load_to_lds(g0, g1, z4, z4, 0);
#endif
}

#define WMMA_F16(A_, B_, C_) \
  __builtin_amdgcn_wmma_f32_16x16x32_f16(false, (A_), false, (B_), (short)0, (C_), false, false)

// ---------------- generic batched NT GEMM: C = alpha*(A @ B^T) + bias -------
// A: f16 [M,K] row-major (global, register double-buffered fragment quads).
// B: f16 [N,K] row-major, staged per 64-wide k-step into LDS by the TDM
//    (leader wave), double-buffered so DMA of step i+1 overlaps step i WMMAs.
// 256 thr = 8 waves, block tile 256x64, wave tile 32x64: 16 WMMAs per k-step,
// each B fragment feeds two WMMAs (row blocks), amortizing LDS latency.
// MODE 0: f16 out [M,N]; MODE 1: f16 out transposed per 1024-row sub-batch
// (row m -> (b=m>>10, s=m&1023), C[b*tBatch + n*1024 + s]); MODE 2: f32 out.
template <int MODE>
__global__ __launch_bounds__(256) void gemm_nt(
    const h16* __restrict__ A, const h16* __restrict__ B,
    const float* __restrict__ bias, void* __restrict__ C,
    int K, int lda, int ldb, int ldc,
    long sA, long sB, long sC, float alpha, long tBatch) {
  __shared__ __align__(16) h16 bsm[2][64 * 64];

  const int lane = threadIdx.x & 31;
  const int bz   = blockIdx.z;
  const h16* Ab = A + (size_t)bz * sA;
  const h16* Bb = B + (size_t)bz * sB;
  const int m0  = blockIdx.x * 256 + (threadIdx.x >> 5) * 32;
  const int n0  = blockIdx.y * 64;
  const int row = lane & 15;
  const int kb2 = (lane & 16) ? 16 : 0;

  // Uniform (SGPR) leader-wave predicate -> scalar branch, no exec juggling.
  const bool leader = (__builtin_amdgcn_readfirstlane((int)threadIdx.x) >> 5) == 0;

  const h16* Btile = Bb + (size_t)n0 * ldb;  // rows n0..n0+63

  if (leader) {
    tdm_load_tile((unsigned)(size_t)&bsm[0][0], Btile, ldb);
    __builtin_amdgcn_s_wait_tensorcnt(0);
  }
  wg_barrier();

  v8f acc[8] = {};
  const int arow0 = m0 + row;        // row block 0
  const int arow1 = arow0 + 16;      // row block 1
  v16h a00 = frag_a(Ab, lda, arow0, 0, lane);
  v16h a01 = frag_a(Ab, lda, arow0, 32, lane);
  v16h a10 = frag_a(Ab, lda, arow1, 0, lane);
  v16h a11 = frag_a(Ab, lda, arow1, 32, lane);

  const int nsteps = K >> 6;  // K % 64 == 0 for every GEMM in this model
  for (int i = 0; i < nsteps; ++i) {
    const int cur = i & 1;
    if (leader && i + 1 < nsteps) {
      tdm_load_tile((unsigned)(size_t)&bsm[1 - cur][0], Btile + (i + 1) * 64, ldb);
    }
    // Prefetch next A fragment quad (stays in flight across the barrier).
    const int kn = (i + 1 < nsteps) ? (i + 1) * 64 : 0;
    v16h an00 = frag_a(Ab, lda, arow0, kn, lane);
    v16h an01 = frag_a(Ab, lda, arow0, kn + 32, lane);
    v16h an10 = frag_a(Ab, lda, arow1, kn, lane);
    v16h an11 = frag_a(Ab, lda, arow1, kn + 32, lane);

    const h16* base = &bsm[cur][row * 64 + kb2];
    // k-chunk 0: each B fragment feeds both row blocks.
    {
      v16h b0 = frag_b_lds(base);
      v16h b1 = frag_b_lds(base + 1024);
      v16h b2 = frag_b_lds(base + 2048);
      v16h b3 = frag_b_lds(base + 3072);
      acc[0] = WMMA_F16(a00, b0, acc[0]);
      acc[4] = WMMA_F16(a10, b0, acc[4]);
      acc[1] = WMMA_F16(a00, b1, acc[1]);
      acc[5] = WMMA_F16(a10, b1, acc[5]);
      acc[2] = WMMA_F16(a00, b2, acc[2]);
      acc[6] = WMMA_F16(a10, b2, acc[6]);
      acc[3] = WMMA_F16(a00, b3, acc[3]);
      acc[7] = WMMA_F16(a10, b3, acc[7]);
    }
    // k-chunk 1 (K offset +32 within the LDS tile).
    {
      v16h b0 = frag_b_lds(base + 32);
      v16h b1 = frag_b_lds(base + 1024 + 32);
      v16h b2 = frag_b_lds(base + 2048 + 32);
      v16h b3 = frag_b_lds(base + 3072 + 32);
      acc[0] = WMMA_F16(a01, b0, acc[0]);
      acc[4] = WMMA_F16(a11, b0, acc[4]);
      acc[1] = WMMA_F16(a01, b1, acc[1]);
      acc[5] = WMMA_F16(a11, b1, acc[5]);
      acc[2] = WMMA_F16(a01, b2, acc[2]);
      acc[6] = WMMA_F16(a11, b2, acc[6]);
      acc[3] = WMMA_F16(a01, b3, acc[3]);
      acc[7] = WMMA_F16(a11, b3, acc[7]);
    }
    a00 = an00; a01 = an01; a10 = an10; a11 = an11;
    if (leader) __builtin_amdgcn_s_wait_tensorcnt(0);
    wg_barrier();
  }

  // C/D layout: VGPR r -> M = base + r (+8 for lanes 16-31), N = n0 + (lane&15)
  const int lanehi = (lane >> 4) << 3;
#pragma unroll
  for (int rb = 0; rb < 2; ++rb) {
    const int mb = m0 + rb * 16 + lanehi;
#pragma unroll
    for (int t = 0; t < 4; ++t) {
      const int nn = n0 + t * 16 + row;
      const float bv = bias ? bias[nn] : 0.0f;
#pragma unroll
      for (int r = 0; r < 8; ++r) {
        float c = acc[rb * 4 + t][r] * alpha + bv;
        int mm = mb + r;
        if (MODE == 0) {
          ((h16*)C)[(size_t)bz * sC + (size_t)mm * ldc + nn] = (h16)c;
        } else if (MODE == 1) {
          int b2 = mm >> 10, sr = mm & 1023;
          ((h16*)C)[(size_t)b2 * tBatch + (size_t)nn * 1024 + sr] = (h16)c;
        } else {
          ((float*)C)[(size_t)bz * sC + (size_t)mm * ldc + nn] = c;
        }
      }
    }
  }
}

// ---------------- row softmax over 1024 f16 elements, in place --------------
// One wave per row; each lane owns 32 contiguous halves (4x b128 each way).
__global__ __launch_bounds__(256) void softmax1024(h16* __restrict__ P) {
  const long row = (long)blockIdx.x * 8 + (threadIdx.x >> 5);
  const int lane = threadIdx.x & 31;
  h16* p = P + row * 1024 + lane * 32;
  v8h c[4];
#pragma unroll
  for (int j = 0; j < 4; ++j) c[j] = *(const v8h*)(p + j * 8);
  float v[32];
  float m = -3.0e38f;
#pragma unroll
  for (int j = 0; j < 4; ++j)
#pragma unroll
    for (int e = 0; e < 8; ++e) { v[j * 8 + e] = (float)c[j][e]; m = fmaxf(m, v[j * 8 + e]); }
#pragma unroll
  for (int off = 16; off > 0; off >>= 1) m = fmaxf(m, __shfl_xor(m, off, 32));
  float s = 0.0f;
#pragma unroll
  for (int i = 0; i < 32; ++i) { v[i] = __expf(v[i] - m); s += v[i]; }
#pragma unroll
  for (int off = 16; off > 0; off >>= 1) s += __shfl_xor(s, off, 32);
  const float inv = 1.0f / s;
#pragma unroll
  for (int j = 0; j < 4; ++j) {
    v8h o;
#pragma unroll
    for (int e = 0; e < 8; ++e) o[e] = (h16)(v[j * 8 + e] * inv);
    *(v8h*)(p + j * 8) = o;
  }
}

// ---------------- helpers ----------------------------------------------------
__global__ __launch_bounds__(256) void cvt_f32_f16(const float* __restrict__ in,
                                                   h16* __restrict__ out, long n) {
  long i = ((long)blockIdx.x * 256 + threadIdx.x) * 4;
  if (i + 3 < n) {
    float4 f = *(const float4*)(in + i);
    v4h o = {(h16)f.x, (h16)f.y, (h16)f.z, (h16)f.w};
    *(v4h*)(out + i) = o;
  }
}

__global__ __launch_bounds__(256) void transpose_w(const float* __restrict__ W,
                                                   h16* __restrict__ Wt, int R, int C) {
  int idx = blockIdx.x * 256 + threadIdx.x;
  if (idx >= R * C) return;
  int r = idx / C, c = idx - r * C;
  Wt[(size_t)c * R + r] = (h16)W[idx];
}

// ---------------- driver ------------------------------------------------------
extern "C" void kernel_launch(void* const* d_in, const int* in_sizes, int n_in,
                              void* d_out, int out_size, void* d_ws, size_t ws_size,
                              hipStream_t stream) {
  const float* x   = (const float*)d_in[0];
  const float* enc = (const float*)d_in[1];
  const float* Wq  = (const float*)d_in[2];
  const float* bq  = (const float*)d_in[3];
  const float* Wk  = (const float*)d_in[4];
  const float* bk  = (const float*)d_in[5];
  const float* Wv  = (const float*)d_in[6];
  const float* bv  = (const float*)d_in[7];
  const float* Wp  = (const float*)d_in[8];
  const float* bp  = (const float*)d_in[9];

  const long XE = (long)NB * NT * NE;  // 6291456
  const long WW = (long)NE * NH;       // 589824
  const long QN = (long)NB * NT * NH;  // 6291456
  const long SN = (long)NB * NT * NS;  // 8388608

  h16* w   = (h16*)d_ws;
  h16* x16 = w;  w += XE;
  h16* e16 = w;  w += XE;
  h16* Wqt = w;  w += WW;
  h16* Wkt = w;  w += WW;
  h16* Wvt = w;  w += WW;
  h16* Wpt = w;  w += WW;
  h16* Qf  = w;  w += QN;
  h16* Kf  = w;  w += QN;
  h16* Vt  = w;  w += QN;   // [B][H][S]
  h16* Sc  = w;  w += SN;   // scores -> probs in place
  h16* Of  = w;  w += QN;

  const dim3 blk(256);
  const float rsH = 0.03608439182435161f;  // 1/sqrt(768)

  cvt_f32_f16<<<(XE / 4 + 255) / 256, blk, 0, stream>>>(x,   x16, XE);
  cvt_f32_f16<<<(XE / 4 + 255) / 256, blk, 0, stream>>>(enc, e16, XE);
  transpose_w<<<(WW + 255) / 256, blk, 0, stream>>>(Wq, Wqt, NE, NH);
  transpose_w<<<(WW + 255) / 256, blk, 0, stream>>>(Wk, Wkt, NE, NH);
  transpose_w<<<(WW + 255) / 256, blk, 0, stream>>>(Wv, Wvt, NE, NH);
  transpose_w<<<(WW + 255) / 256, blk, 0, stream>>>(Wp, Wpt, NH, NE);

  // Q = x @ Wq + bq   (M=8192, N=768, K=768)
  gemm_nt<0><<<dim3(32, 12, 1), blk, 0, stream>>>(
      x16, Wqt, bq, Qf, NE, NE, NE, NH, 0, 0, 0, 1.0f, 0);
  // K = enc @ Wk + bk
  gemm_nt<0><<<dim3(32, 12, 1), blk, 0, stream>>>(
      e16, Wkt, bk, Kf, NE, NE, NE, NH, 0, 0, 0, 1.0f, 0);
  // V = enc @ Wv + bv, stored transposed per batch: Vt[b][h][s]
  gemm_nt<1><<<dim3(32, 12, 1), blk, 0, stream>>>(
      e16, Wvt, bv, Vt, NE, NE, NE, 0, 0, 0, 0, 1.0f, (long)NH * NS);
  // scores = (Q @ K^T) * 1/sqrt(H)   batched (M=T, N=S, K=H)
  gemm_nt<0><<<dim3(4, 16, NB), blk, 0, stream>>>(
      Qf, Kf, nullptr, Sc, NH, NH, NH, NS,
      (long)NT * NH, (long)NS * NH, (long)NT * NS, rsH, 0);
  // softmax rows (in place)
  softmax1024<<<SN / 8192, blk, 0, stream>>>(Sc);
  // O = P @ V    batched (M=T, N=H, K=S), B = Vt so it's NT
  gemm_nt<0><<<dim3(4, 12, NB), blk, 0, stream>>>(
      Sc, Vt, nullptr, Of, NS, NS, NS, NH,
      (long)NT * NS, (long)NH * NS, (long)NT * NH, 1.0f, 0);
  // out = O @ Wp + bp  (f32 output)
  gemm_nt<2><<<dim3(32, 12, 1), blk, 0, stream>>>(
      Of, Wpt, bp, d_out, NH, NH, NH, NE, 0, 0, 0, 1.0f, 0);
}
